// CLDNN_69380901700295
// MI455X (gfx1250) — matrix operand order repaired and checked by reading, not compiled, not run on hardware
//
#include <hip/hip_runtime.h>
#include <math.h>

// ---------------- types ----------------
typedef __attribute__((ext_vector_type(16))) _Float16 v16h;
typedef __attribute__((ext_vector_type(8)))  _Float16 v8h;
typedef __attribute__((ext_vector_type(8)))  float    v8f;

union V16H { v16h v; v8h h[2]; };

#define HIDN 512
#define TT 64
#define BB 32

// ---------------- WMMA helpers (CDNA5 gfx1250, wave32) ----------------
__device__ __forceinline__ v8f wmma_f16(v16h a, v16h b, v8f c) {
  // D = A(16x32 f16) * B(32x16 f16) + C(16x16 f32)
  return __builtin_amdgcn_wmma_f32_16x16x32_f16(
      false, a, false, b, (short)0, c, false, false);
}

// A fragment: 16x32 tile, row-major source, row r = lane&15, hf = lane>>4.
// ISA layout: hf=0 -> K {k0..k0+7} in elems 0..7, {k0+16..k0+23} in elems 8..15
//             hf=1 -> K {k0+8..}  and {k0+24..}
__device__ __forceinline__ v16h load_a_frag(const _Float16* base, int stride,
                                            int k0, int r, int hf) {
  const _Float16* p = base + (size_t)r * stride + k0 + hf * 8;
  V16H u;
  u.h[0] = *(const v8h*)(p);
  u.h[1] = *(const v8h*)(p + 16);
  return u.v;
}

// B fragment: 32x16 tile of B = W^T, sourced from row-major W[n][k].
// Lane holds column n = lane&15; hf=0 -> K {k0..k0+7, k0+16..k0+23}, hf=1 -> K {k0+8..k0+15, k0+24..k0+31}.
__device__ __forceinline__ v16h load_b_frag(const _Float16* base, int stride,
                                            int k0, int n, int hf) {
  const _Float16* p = base + (size_t)n * stride + k0 + hf * 8;
  V16H u;
  u.h[0] = *(const v8h*)(p);
  u.h[1] = *(const v8h*)(p + 16);
  return u.v;
}

__device__ __forceinline__ float sigm(float x) { return 1.0f / (1.0f + __expf(-x)); }

// ---------------- small utility kernels ----------------
__global__ void f32_to_f16_kernel(const float* __restrict__ s,
                                  _Float16* __restrict__ d, int n) {
  int i = blockIdx.x * blockDim.x + threadIdx.x;
  if (i < n) d[i] = (_Float16)s[i];
}

__global__ void pad_fcw_kernel(const float* __restrict__ s,  // [256][30]
                               _Float16* __restrict__ d) {   // [256][32]
  int i = blockIdx.x * blockDim.x + threadIdx.x;
  if (i >= 256 * 32) return;
  int row = i >> 5, k = i & 31;
  d[i] = (k < 30) ? (_Float16)s[row * 30 + k] : (_Float16)0.0f;
}

__global__ void zero_f32_kernel(float* p, int n) {
  int i = blockIdx.x * blockDim.x + threadIdx.x;
  if (i < n) p[i] = 0.0f;
}
__global__ void zero_f16_kernel(_Float16* p, int n) {
  int i = blockIdx.x * blockDim.x + threadIdx.x;
  if (i < n) p[i] = (_Float16)0.0f;
}

// ---------------- CNN (fp32, fused conv+relu+pool) ----------------
// conv1 3x3 pad1 (3->10) + relu + maxpool2 : out [nimg][10][32][32]
__global__ void conv1_pool_kernel(const float* __restrict__ x,
                                  const float* __restrict__ w,
                                  const float* __restrict__ b,
                                  float* __restrict__ out, int img0, int nimg) {
  int idx = blockIdx.x * blockDim.x + threadIdx.x;
  int total = nimg * 10 * 32 * 32;
  if (idx >= total) return;
  int pw = idx & 31;
  int ph = (idx >> 5) & 31;
  int rest = idx >> 10;
  int oc = rest % 10;
  int li = rest / 10;
  const float* xi = x + (size_t)(img0 + li) * 3 * 4096;
  float m = -1e30f;
  for (int dy = 0; dy < 2; ++dy)
    for (int dx = 0; dx < 2; ++dx) {
      int oy = 2 * ph + dy, ox = 2 * pw + dx;
      float s = b[oc];
      for (int ic = 0; ic < 3; ++ic)
        for (int ky = 0; ky < 3; ++ky) {
          int y = oy + ky - 1;
          if ((unsigned)y >= 64u) continue;
          for (int kx = 0; kx < 3; ++kx) {
            int xx = ox + kx - 1;
            if ((unsigned)xx >= 64u) continue;
            s += xi[ic * 4096 + y * 64 + xx] * w[((oc * 3 + ic) * 3 + ky) * 3 + kx];
          }
        }
      m = fmaxf(m, s);
    }
  out[idx] = fmaxf(m, 0.0f);  // relu(max) == max(relu)
}

// conv2 2x2 valid (10->20) + relu + maxpool2 : in [nimg][10][32][32] -> out [nimg][20][15][15]
__global__ void conv2_pool_kernel(const float* __restrict__ in,
                                  const float* __restrict__ w,
                                  const float* __restrict__ b,
                                  float* __restrict__ out, int nimg) {
  int idx = blockIdx.x * blockDim.x + threadIdx.x;
  int total = nimg * 20 * 15 * 15;
  if (idx >= total) return;
  int pw = idx % 15;
  int ph = (idx / 15) % 15;
  int oc = (idx / 225) % 20;
  int li = idx / 4500;
  const float* pin = in + (size_t)li * 10 * 1024;
  float m = -1e30f;
  for (int dy = 0; dy < 2; ++dy)
    for (int dx = 0; dx < 2; ++dx) {
      int oy = 2 * ph + dy, ox = 2 * pw + dx;
      float s = b[oc];
      for (int ic = 0; ic < 10; ++ic) {
        const float* pp = pin + ic * 1024 + oy * 32 + ox;
        const float* ww = w + (oc * 10 + ic) * 4;
        s += pp[0] * ww[0] + pp[1] * ww[1] + pp[32] * ww[2] + pp[33] * ww[3];
      }
      m = fmaxf(m, s);
    }
  out[idx] = fmaxf(m, 0.0f);
}

// conv3 2x2 valid (20->30) + relu + global avg pool : out feat16 [2048][32] (K-padded)
__global__ void conv3_avg_kernel(const float* __restrict__ in,
                                 const float* __restrict__ w,
                                 const float* __restrict__ b,
                                 _Float16* __restrict__ feat, int img0, int nimg) {
  int idx = blockIdx.x * blockDim.x + threadIdx.x;
  if (idx >= nimg * 32) return;
  int oc = idx & 31, li = idx >> 5;
  size_t o = (size_t)(img0 + li) * 32 + oc;
  if (oc >= 30) { feat[o] = (_Float16)0.0f; return; }
  const float* pin = in + (size_t)li * 20 * 225;
  float acc = 0.0f;
  for (int y = 0; y < 14; ++y)
    for (int x = 0; x < 14; ++x) {
      float s = b[oc];
      for (int ic = 0; ic < 20; ++ic) {
        const float* pp = pin + ic * 225 + y * 15 + x;
        const float* ww = w + (oc * 20 + ic) * 4;
        s += pp[0] * ww[0] + pp[1] * ww[1] + pp[15] * ww[2] + pp[16] * ww[3];
      }
      acc += fmaxf(s, 0.0f);
    }
  feat[o] = (_Float16)(acc * (1.0f / 196.0f));
}

// ---------------- cnn_fc: [2048x32] @ [256x32]^T via WMMA, relu, emit f16 ----------------
__global__ __launch_bounds__(256) void cnnfc_kernel(const _Float16* __restrict__ feat,
                                                    const _Float16* __restrict__ w,
                                                    const float* __restrict__ b,
                                                    _Float16* __restrict__ out) {
  int wave = threadIdx.x >> 5;
  int lane = threadIdx.x & 31;
  int r = lane & 15, hf = lane >> 4;
  int m0 = blockIdx.x * 16;
  for (int nt = wave; nt < 16; nt += 8) {  // uniform per wave
    int n0 = nt * 16;
    v16h a = load_a_frag(feat + (size_t)m0 * 32, 32, 0, r, hf);
    v16h bb = load_b_frag(w + (size_t)n0 * 32, 32, 0, r, hf);
    v8f c = {0, 0, 0, 0, 0, 0, 0, 0};
    c = wmma_f16(a, bb, c);
    float bias = b[n0 + r];
    #pragma unroll
    for (int i = 0; i < 8; ++i) {
      int row = m0 + i + 8 * hf;
      float v = fmaxf(c[i] + bias, 0.0f);
      out[(size_t)row * 256 + n0 + r] = (_Float16)v;
    }
  }
}

// ---------------- LSTM peephole cell step (WMMA gates GEMM + epilogue) ----------------
// gates[32 x 2048] = A[32 x Kin] @ Wih^T + Hin[32 x 512] @ Whh^T (+ biases)
// grid = 16 blocks x 64 threads (2 waves); each wave owns 16 hidden columns
// and computes all 4 gate groups for them -> fully local state update.
__global__ __launch_bounds__(64) void lstm_cell_kernel(
    const _Float16* A, int aStride, int Kin,
    const _Float16* __restrict__ Wih,
    const _Float16* Hin,
    const _Float16* __restrict__ Whh,
    const float* __restrict__ bih, const float* __restrict__ bhh,
    const float* __restrict__ pi, const float* __restrict__ pf,
    const float* __restrict__ po,
    float* __restrict__ cx, float* __restrict__ hx32,
    _Float16* __restrict__ Hout, _Float16* seqOut) {
  const int wave = threadIdx.x >> 5;
  const int lane = threadIdx.x & 31;
  const int r = lane & 15, hf = lane >> 4;
  const int h0 = blockIdx.x * 32 + wave * 16;  // hidden-column base (0..511)

  v8f acc[4][2];
  #pragma unroll
  for (int g = 0; g < 4; ++g)
    #pragma unroll
    for (int mt = 0; mt < 2; ++mt) acc[g][mt] = (v8f){0, 0, 0, 0, 0, 0, 0, 0};

  // input contribution: A @ Wih^T
  for (int k0 = 0; k0 < Kin; k0 += 32) {
    v16h a0 = load_a_frag(A, aStride, k0, r, hf);
    v16h a1 = load_a_frag(A + (size_t)16 * aStride, aStride, k0, r, hf);
    #pragma unroll
    for (int g = 0; g < 4; ++g) {
      v16h bf = load_b_frag(Wih + (size_t)(g * HIDN + h0) * Kin, Kin, k0, r, hf);
      acc[g][0] = wmma_f16(a0, bf, acc[g][0]);
      acc[g][1] = wmma_f16(a1, bf, acc[g][1]);
    }
  }
  // recurrent contribution: Hin @ Whh^T
  for (int k0 = 0; k0 < HIDN; k0 += 32) {
    v16h a0 = load_a_frag(Hin, HIDN, k0, r, hf);
    v16h a1 = load_a_frag(Hin + 16 * HIDN, HIDN, k0, r, hf);
    #pragma unroll
    for (int g = 0; g < 4; ++g) {
      v16h bf = load_b_frag(Whh + (size_t)(g * HIDN + h0) * HIDN, HIDN, k0, r, hf);
      acc[g][0] = wmma_f16(a0, bf, acc[g][0]);
      acc[g][1] = wmma_f16(a1, bf, acc[g][1]);
    }
  }

  const int hcol = h0 + r;
  float bsum[4];
  #pragma unroll
  for (int g = 0; g < 4; ++g) bsum[g] = bih[g * HIDN + hcol] + bhh[g * HIDN + hcol];
  const float Pi = pi[hcol], Pf = pf[hcol], Po = po[hcol];

  #pragma unroll
  for (int mt = 0; mt < 2; ++mt)
    #pragma unroll
    for (int i = 0; i < 8; ++i) {
      int row = mt * 16 + i + 8 * hf;         // batch row 0..31 (C-layout)
      size_t idx = (size_t)row * HIDN + hcol;
      float c0 = cx[idx];
      float gi = acc[0][mt][i] + bsum[0];
      float gf = acc[1][mt][i] + bsum[1];
      float gg = acc[2][mt][i] + bsum[2];
      float go = acc[3][mt][i] + bsum[3];
      float ig = sigm(gi + Pi * c0);
      float fg = sigm(gf + Pf * c0);
      float cn = fg * c0 + ig * tanhf(gg);
      float og = sigm(go + Po * cn);
      float hn = og * tanhf(cn);
      cx[idx] = cn;
      hx32[idx] = hn;
      _Float16 hh = (_Float16)hn;
      Hout[idx] = hh;
      if (seqOut) seqOut[idx] = hh;
    }
}

// ---------------- DNN head (tiny, fp32) ----------------
__global__ void fc_kernel(const float* __restrict__ in, const float* __restrict__ w,
                          const float* __restrict__ b, float* __restrict__ out,
                          int K, int N, int act /*0=none 1=relu 2=sigmoid*/) {
  int idx = blockIdx.x * blockDim.x + threadIdx.x;
  if (idx >= BB * N) return;
  int n = idx % N, row = idx / N;
  const float* pin = in + (size_t)row * K;
  const float* pw = w + (size_t)n * K;
  float s = b[n];
  for (int k = 0; k < K; ++k) s += pin[k] * pw[k];
  if (act == 1) s = fmaxf(s, 0.0f);
  else if (act == 2) s = 1.0f / (1.0f + expf(-s));
  out[idx] = s;
}

// ---------------- host orchestration ----------------
extern "C" void kernel_launch(void* const* d_in, const int* in_sizes, int n_in,
                              void* d_out, int out_size, void* d_ws, size_t ws_size,
                              hipStream_t stream) {
  (void)out_size; (void)ws_size;
  if (n_in < 43) return;

  const float* x   = (const float*)d_in[0];
  const float* c1w = (const float*)d_in[1];  const float* c1b = (const float*)d_in[2];
  const float* c2w = (const float*)d_in[3];  const float* c2b = (const float*)d_in[4];
  const float* c3w = (const float*)d_in[5];  const float* c3b = (const float*)d_in[6];
  const float* fcw = (const float*)d_in[7];  const float* fcb = (const float*)d_in[8];

  // LSTM leaves: 4 cells x 7 starting at index 9. Dict key order may be
  // insertion (Wih,Whh,bih,bhh,pi,pf,po) or alphabetical (Whh,Wih,bhh,bih,pf,pi,po).
  // Disambiguate with cell(0,0): Wih there is 2048x256, Whh is 2048x512.
  bool alpha = (in_sizes[9] == 2048 * 512);
  const float *Wih[4], *Whh[4], *bih[4], *bhh[4], *Ppi[4], *Ppf[4], *Ppo[4];
  int Kin[4] = {256, 512, 512, 512};
  for (int c = 0; c < 4; ++c) {
    int base = 9 + 7 * c;
    if (alpha) {
      Whh[c] = (const float*)d_in[base + 0]; Wih[c] = (const float*)d_in[base + 1];
      bhh[c] = (const float*)d_in[base + 2]; bih[c] = (const float*)d_in[base + 3];
      Ppf[c] = (const float*)d_in[base + 4]; Ppi[c] = (const float*)d_in[base + 5];
      Ppo[c] = (const float*)d_in[base + 6];
    } else {
      Wih[c] = (const float*)d_in[base + 0]; Whh[c] = (const float*)d_in[base + 1];
      bih[c] = (const float*)d_in[base + 2]; bhh[c] = (const float*)d_in[base + 3];
      Ppi[c] = (const float*)d_in[base + 4]; Ppf[c] = (const float*)d_in[base + 5];
      Ppo[c] = (const float*)d_in[base + 6];
    }
  }
  const float* f1w = (const float*)d_in[37]; const float* f1b = (const float*)d_in[38];
  const float* f2w = (const float*)d_in[39]; const float* f2b = (const float*)d_in[40];
  const float* f3w = (const float*)d_in[41]; const float* f3b = (const float*)d_in[42];

  // bump allocator over workspace (peak ~40 MB)
  char* wsb = (char*)d_ws;
  size_t off = 0;
  auto alloc = [&](size_t bytes) -> void* {
    off = (off + 255) & ~(size_t)255;
    void* p = wsb + off;
    off += bytes;
    return p;
  };

  _Float16* Wih16[4]; _Float16* Whh16[4];
  for (int c = 0; c < 4; ++c) {
    Wih16[c] = (_Float16*)alloc((size_t)2048 * Kin[c] * 2);
    Whh16[c] = (_Float16*)alloc((size_t)2048 * HIDN * 2);
  }
  _Float16* fcw16  = (_Float16*)alloc(256 * 32 * 2);
  _Float16* feat16 = (_Float16*)alloc((size_t)2048 * 32 * 2);
  _Float16* cnn16  = (_Float16*)alloc((size_t)2048 * 256 * 2);  // [img=b*64+t][256]
  _Float16* seq0   = (_Float16*)alloc((size_t)TT * BB * HIDN * 2);
  _Float16* hbuf[2];
  hbuf[0] = (_Float16*)alloc((size_t)BB * HIDN * 2);
  hbuf[1] = (_Float16*)alloc((size_t)BB * HIDN * 2);
  float* cx    = (float*)alloc((size_t)BB * HIDN * 4);
  float* hx32  = (float*)alloc((size_t)BB * HIDN * 4);
  float* pool1 = (float*)alloc((size_t)256 * 10 * 32 * 32 * 4);
  float* pool2 = (float*)alloc((size_t)256 * 20 * 15 * 15 * 4);
  float* t1    = (float*)alloc((size_t)BB * 200 * 4);
  float* t2    = (float*)alloc((size_t)BB * 100 * 4);

  // weight conversion to f16
  for (int c = 0; c < 4; ++c) {
    int n1 = 2048 * Kin[c];
    f32_to_f16_kernel<<<(n1 + 255) / 256, 256, 0, stream>>>(Wih[c], Wih16[c], n1);
    int n2 = 2048 * HIDN;
    f32_to_f16_kernel<<<(n2 + 255) / 256, 256, 0, stream>>>(Whh[c], Whh16[c], n2);
  }
  pad_fcw_kernel<<<(256 * 32 + 255) / 256, 256, 0, stream>>>(fcw, fcw16);

  // CNN over 8 chunks of 256 images
  for (int ch = 0; ch < 8; ++ch) {
    int img0 = ch * 256, nimg = 256;
    int t1n = nimg * 10 * 32 * 32;
    conv1_pool_kernel<<<(t1n + 255) / 256, 256, 0, stream>>>(x, c1w, c1b, pool1, img0, nimg);
    int t2n = nimg * 20 * 15 * 15;
    conv2_pool_kernel<<<(t2n + 255) / 256, 256, 0, stream>>>(pool1, c2w, c2b, pool2, nimg);
    int t3n = nimg * 32;
    conv3_avg_kernel<<<(t3n + 255) / 256, 256, 0, stream>>>(pool2, c3w, c3b, feat16, img0, nimg);
  }
  cnnfc_kernel<<<128, 256, 0, stream>>>(feat16, fcw16, fcb, cnn16);

  // LSTM: 2 blocks x 64 steps x 2 stacked cells, shared (hx,cx) state
  int cur = 0;
  for (int lb = 0; lb < 2; ++lb) {
    zero_f32_kernel<<<(BB * HIDN + 255) / 256, 256, 0, stream>>>(cx, BB * HIDN);
    zero_f16_kernel<<<(BB * HIDN + 255) / 256, 256, 0, stream>>>(hbuf[cur], BB * HIDN);
    for (int t = 0; t < TT; ++t) {
      for (int c = 0; c < 2; ++c) {
        int ci = lb * 2 + c;
        const _Float16* Hin = hbuf[cur];
        _Float16* Hout = hbuf[cur ^ 1];
        const _Float16* A;
        int aStride;
        if (c == 0) {
          if (lb == 0) { A = cnn16 + (size_t)t * 256; aStride = TT * 256; }  // row b at (b*64+t)*256
          else         { A = seq0 + (size_t)t * BB * HIDN; aStride = HIDN; }
        } else {
          A = Hin; aStride = HIDN;  // stacked cell consumes current hidden state
        }
        _Float16* seqOut = (c == 1 && lb == 0) ? (seq0 + (size_t)t * BB * HIDN) : nullptr;
        lstm_cell_kernel<<<16, 64, 0, stream>>>(
            A, aStride, Kin[ci], Wih16[ci], Hin, Whh16[ci],
            bih[ci], bhh[ci], Ppi[ci], Ppf[ci], Ppo[ci],
            cx, hx32, Hout, seqOut);
        cur ^= 1;
      }
    }
  }

  // DNN head on last hidden state
  fc_kernel<<<(BB * 200 + 255) / 256, 256, 0, stream>>>(hx32, f1w, f1b, t1, HIDN, 200, 1);
  fc_kernel<<<(BB * 100 + 255) / 256, 256, 0, stream>>>(t1, f2w, f2b, t2, 200, 100, 1);
  fc_kernel<<<(BB * 10 + 255) / 256, 256, 0, stream>>>(t2, f3w, f3b, (float*)d_out, 100, 10, 2);
}
